// DualAttention2d_59390807769339
// MI455X (gfx1250) — matrix operand order repaired and checked by reading, not compiled, not run on hardware
//
#include <hip/hip_runtime.h>
#include <math.h>

#define DEV __device__ __forceinline__

typedef __attribute__((ext_vector_type(16))) __bf16         v16bf;
typedef __attribute__((ext_vector_type(16))) unsigned short v16u;
typedef __attribute__((ext_vector_type(8)))  float          v8f;

static constexpr int   Bn = 4, Cc = 512, Ss = 4096, CI = 64;
static constexpr int   Hp = 66, SP = 66 * 66; // zero-padded spatial (1-px halo)
static constexpr float EPSf = 1e-5f;

// scheduling fence: VALU/SALU may cross, VMEM & WMMA stay in program-order groups
DEV void sfence() { __builtin_amdgcn_sched_barrier(0x6); }

// ---------------- bf16 helpers ----------------
DEV unsigned short f2bf(float f) {
  unsigned u = __float_as_uint(f);
  u += 0x7FFFu + ((u >> 16) & 1u);
  return (unsigned short)(u >> 16);
}
DEV float bf2f(unsigned short h) { return __uint_as_float(((unsigned)h) << 16); }

// flat spatial n (0..4095) -> padded offset within one image
DEV int pad_off(int n) { return ((n >> 6) + 1) * Hp + (n & 63) + 1; }

DEV v16u frag_zero() {
  union { uint4 q[2]; v16u v; } u;
  u.q[0] = make_uint4(0, 0, 0, 0);
  u.q[1] = make_uint4(0, 0, 0, 0);
  return u.v;
}
DEV v8f vzero() {
  union { float f[8]; v8f v; } u;
#pragma unroll
  for (int i = 0; i < 8; ++i) u.f[i] = 0.f;
  return u.v;
}

// A-matrix fragment (16x32 bf16), A row-major [row][k], stride ld.
// lane<16: row=lane, K {0..7,16..23}; lane>=16: row=lane-16, K {8..15,24..31}
DEV v16u frag_ldA(const unsigned short* base, int ld, int lane) {
  int r = lane & 15, ko = (lane >> 4) << 3;
  const unsigned short* p = base + (size_t)r * ld + ko;
  union { uint4 q[2]; v16u v; } u;
  u.q[0] = *(const uint4*)(p);
  u.q[1] = *(const uint4*)(p + 16);
  return u.v;
}
// B-matrix fragment (32x16 bf16) given Bt row-major [n][k], stride ld.
// lane<16: col=lane, K {0..7,16..23}; lane>=16: col=lane-16, K {8..15,24..31}
DEV v16u frag_ldB(const unsigned short* base, int ld, int lane) {
  int r = lane & 15, ko = (lane >> 4) << 3;
  const unsigned short* p = base + (size_t)r * ld + ko;
  union { uint4 q[2]; v16u v; } u;
  u.q[0] = *(const uint4*)(p);
  u.q[1] = *(const uint4*)(p + 16);
  return u.v;
}

DEV v8f wmma32(v16u a, v16u b, v8f c) {
  return __builtin_amdgcn_wmma_f32_16x16x32_bf16(
      false, __builtin_bit_cast(v16bf, a),
      false, __builtin_bit_cast(v16bf, b),
      (short)0, c, false, false);
}

// ---------------- prep kernels ----------------
// x [B][C][S] f32 -> padded feature-last bf16 [B][SP][C]
__global__ void k_cast_x(const float* __restrict__ x, unsigned short* __restrict__ xt) {
  size_t i = (size_t)blockIdx.x * blockDim.x + threadIdx.x;
  size_t total = (size_t)Bn * Ss * Cc;
  if (i >= total) return;
  int c = (int)(i % Cc);
  size_t t = i / Cc;
  int n = (int)(t % Ss), b = (int)(t / Ss);
  xt[((size_t)b * SP + pad_off(n)) * Cc + c] = f2bf(x[((size_t)b * Cc + c) * Ss + n]);
}

__global__ void k_fold_w(const float* __restrict__ w, const float* __restrict__ g,
                         const float* __restrict__ v, unsigned short* __restrict__ w16) {
  size_t i = (size_t)blockIdx.x * blockDim.x + threadIdx.x;
  size_t total = (size_t)9 * Cc * Cc;
  if (i >= total) return;
  int c = (int)(i % Cc), o = (int)((i / Cc) % Cc), t = (int)(i / ((size_t)Cc * Cc));
  float s = g[o] * rsqrtf(v[o] + EPSf);
  w16[i] = f2bf(w[((size_t)o * Cc + c) * 9 + t] * s);
}

__global__ void k_fold_b(const float* __restrict__ bb, const float* __restrict__ m,
                         const float* __restrict__ g, const float* __restrict__ v,
                         float* __restrict__ biasf) {
  int o = blockIdx.x * blockDim.x + threadIdx.x;
  if (o >= Cc) return;
  float s = g[o] * rsqrtf(v[o] + EPSf);
  biasf[o] = bb[o] - m[o] * s;
}

__global__ void k_cast(const float* __restrict__ in, unsigned short* __restrict__ out, int n) {
  int i = blockIdx.x * blockDim.x + threadIdx.x;
  if (i < n) out[i] = f2bf(in[i]);
}

// ---------------- implicit-GEMM conv3x3 + BN(+bias) + ReLU ----------------
// xt: padded feature-last bf16 [B][SP][Cin]. w16: [9][Cout][Cin] bf16 (BN folded).
// Block 256 = 8 waves; tile 128(M=Cout) x 128(N=spatial); wave = 64x32.
// Flattened 144-step (9 taps x 16 K-chunks) loop, ping-pong double buffering.
__global__ __launch_bounds__(256) void k_conv_gemm(
    const unsigned short* __restrict__ xt, const unsigned short* __restrict__ w16,
    const float* __restrict__ bias,
    unsigned short* out_ft,   // feature-last bf16 [B][S][C] (unpadded) or null
    unsigned short* out_cm,   // channel-major bf16 [B][C][S] or null
    float* out_f32, int accum // NCHW f32 [B][C][S] or null; accum: +=
) {
  int b = blockIdx.z;
  int mbase = blockIdx.y * 128, nbase = blockIdx.x * 128;
  int tid = threadIdx.x, wave = tid >> 5, lane = tid & 31;
  int wm = wave >> 2, wn = wave & 3;
  int m0 = mbase + wm * 64, n0 = nbase + wn * 32;
  int r = lane & 15, hi8 = (lane >> 4) << 3;

  // tap-independent padded B-fragment bases (16 columns never cross a row)
  const unsigned short* bbase[2];
#pragma unroll
  for (int nf = 0; nf < 2; ++nf) {
    int n = n0 + nf * 16;
    bbase[nf] = xt + ((size_t)b * SP + pad_off(n)) * Cc;
  }
  const unsigned short* wbase = w16 + (size_t)m0 * Cc;

  v8f acc[4][2];
#pragma unroll
  for (int mf = 0; mf < 4; ++mf)
#pragma unroll
    for (int nf = 0; nf < 2; ++nf) acc[mf][nf] = vzero();

  auto loadA = [&](int s, v16u* a) {
    int t = s >> 4, kc = (s & 15) << 5;
    const unsigned short* wt = wbase + (size_t)t * Cc * Cc + kc;
#pragma unroll
    for (int mf = 0; mf < 4; ++mf)
      a[mf] = frag_ldA(wt + (size_t)(mf * 16) * Cc, Cc, lane);
  };
  auto loadB = [&](int s, v16u* bb) {
    int t = s >> 4, kc = (s & 15) << 5;
    int dh = t / 3 - 1, dw = t - (t / 3) * 3 - 1;
    int off = (dh * Hp + dw) * Cc + kc;
#pragma unroll
    for (int nf = 0; nf < 2; ++nf)
      bb[nf] = frag_ldB(bbase[nf] + off, Cc, lane);
  };
  auto mmac = [&](v16u* a, v16u* bb) {
#pragma unroll
    for (int nf = 0; nf < 2; ++nf)
#pragma unroll
      for (int mf = 0; mf < 4; ++mf)
        acc[mf][nf] = wmma32(a[mf], bb[nf], acc[mf][nf]);
  };

  v16u a0[4], b0f[2], a1[4], b1f[2];
  loadA(0, a0);
  loadB(0, b0f);
  for (int s = 0; s < 142; s += 2) {
    loadA(s + 1, a1);
    loadB(s + 1, b1f);
    sfence();
    mmac(a0, b0f);
    sfence();
    loadA(s + 2, a0);
    loadB(s + 2, b0f);
    sfence();
    mmac(a1, b1f);
    sfence();
  }
  loadA(143, a1);
  loadB(143, b1f);
  sfence();
  mmac(a0, b0f); // step 142
  sfence();
  mmac(a1, b1f); // step 143

#pragma unroll
  for (int mf = 0; mf < 4; ++mf)
#pragma unroll
    for (int nf = 0; nf < 2; ++nf)
#pragma unroll
      for (int e = 0; e < 8; ++e) {
        int row = m0 + mf * 16 + e + hi8;
        int col = n0 + nf * 16 + r;
        float vv = acc[mf][nf][e] + bias[row];
        vv = vv > 0.f ? vv : 0.f;
        if (out_ft)  out_ft[((size_t)b * Ss + col) * Cc + row] = f2bf(vv);
        if (out_cm)  out_cm[((size_t)b * Cc + row) * Ss + col] = f2bf(vv);
        if (out_f32) {
          size_t o = ((size_t)b * Cc + row) * Ss + col;
          if (accum) out_f32[o] += vv; else out_f32[o] = vv;
        }
      }
}

// ---------------- generic batched GEMM: D = A(MxK) * Bt(NxK)^T ----------------
// mode 0: store f32 row-major. mode 1: +bias[col], bf16 row-major and/or
// channel-major scatter. mode 2: D = scale*acc + resid (bf16) row-major
// (out optionally in padded feature-last layout).
// Ping-pong double-buffered K loop (K/32 even for all uses: 16 or 128).
__global__ __launch_bounds__(256) void k_gemm(
    const unsigned short* __restrict__ A, size_t sA, int lda,
    const unsigned short* __restrict__ Bt, size_t sB, int ldb,
    int K, int N,
    const float* __restrict__ bias,
    int mode, const float* __restrict__ scale_p,
    const unsigned short* __restrict__ resid, size_t sR,
    unsigned short* out_bf, size_t sO, int out_padded,
    unsigned short* out_cm, size_t sOcm, int ldcm,
    float* out_f32, size_t sF) {
  int b = blockIdx.z;
  int m0 = blockIdx.y * 128 + (threadIdx.x >> 5) * 16;
  int nb = blockIdx.x * 64;
  int lane = threadIdx.x & 31;
  const unsigned short* Ab = A + (size_t)b * sA + (size_t)m0 * lda;
  const unsigned short* Bb = Bt + (size_t)b * sB + (size_t)nb * ldb;

  v8f acc[4];
#pragma unroll
  for (int nf = 0; nf < 4; ++nf) acc[nf] = vzero();

  auto loadBf = [&](int kc, v16u* bb) {
#pragma unroll
    for (int nf = 0; nf < 4; ++nf)
      bb[nf] = frag_ldB(Bb + (size_t)(nf * 16) * ldb + kc, ldb, lane);
  };
  auto mmac = [&](v16u a, v16u* bb) {
#pragma unroll
    for (int nf = 0; nf < 4; ++nf) acc[nf] = wmma32(a, bb[nf], acc[nf]);
  };

  v16u a0, a1, b0[4], b1[4];
  a0 = frag_ldA(Ab, lda, lane);
  loadBf(0, b0);
  int kc = 0;
  for (; kc + 128 <= K; kc += 64) {
    a1 = frag_ldA(Ab + kc + 32, lda, lane);
    loadBf(kc + 32, b1);
    sfence();
    mmac(a0, b0);
    sfence();
    a0 = frag_ldA(Ab + kc + 64, lda, lane);
    loadBf(kc + 64, b0);
    sfence();
    mmac(a1, b1);
    sfence();
  }
  a1 = frag_ldA(Ab + K - 32, lda, lane);
  loadBf(K - 32, b1);
  sfence();
  mmac(a0, b0); // step K-64
  sfence();
  mmac(a1, b1); // step K-32

  int r = lane & 15, hi8 = (lane >> 4) << 3;
  float scale = (mode == 2) ? scale_p[0] : 0.f;
#pragma unroll
  for (int nf = 0; nf < 4; ++nf)
#pragma unroll
    for (int e = 0; e < 8; ++e) {
      int row = m0 + e + hi8;
      int col = nb + nf * 16 + r;
      float vv = acc[nf][e];
      if (mode == 0) {
        out_f32[(size_t)b * sF + (size_t)row * N + col] = vv;
      } else if (mode == 1) {
        vv += bias[col];
        if (out_bf) out_bf[(size_t)b * sO + (size_t)row * N + col] = f2bf(vv);
        if (out_cm) out_cm[(size_t)b * sOcm + (size_t)col * ldcm + row] = f2bf(vv);
      } else {
        vv = scale * vv + bf2f(resid[(size_t)b * sR + (size_t)row * N + col]);
        size_t orow = out_padded ? (size_t)pad_off(row) : (size_t)row;
        out_bf[(size_t)b * sO + orow * N + col] = f2bf(vv);
      }
    }
}

// ---------------- spatial attention pass A: row max / logsumexp ----------------
__global__ __launch_bounds__(32) void k_attn_stats(
    const unsigned short* __restrict__ qt, const unsigned short* __restrict__ kt,
    float* __restrict__ mrow, float* __restrict__ lrow) {
  int b = blockIdx.y, qb = blockIdx.x * 16;
  int lane = threadIdx.x & 31;
  const unsigned short* qbse = qt + ((size_t)b * Ss + qb) * CI;
  v16u a0 = frag_ldA(qbse, CI, lane);
  v16u a1 = frag_ldA(qbse + 32, CI, lane);
  float mv[8], lv[8];
#pragma unroll
  for (int e = 0; e < 8; ++e) { mv[e] = -3.0e38f; lv[e] = 0.f; }
  for (int sb = 0; sb < Ss; sb += 16) {
    const unsigned short* kb = kt + ((size_t)b * Ss + sb) * CI;
    v16u b0 = frag_ldB(kb, CI, lane);
    v16u b1 = frag_ldB(kb + 32, CI, lane);
    v8f sc = vzero();
    sc = wmma32(a0, b0, sc);
    sc = wmma32(a1, b1, sc);
#pragma unroll
    for (int e = 0; e < 8; ++e) {
      float v = sc[e];
      if (v <= mv[e]) {
        lv[e] += __expf(v - mv[e]);
      } else {
        lv[e] = lv[e] * __expf(mv[e] - v) + 1.f;
        mv[e] = v;
      }
    }
  }
#pragma unroll
  for (int off = 8; off; off >>= 1)
#pragma unroll
    for (int e = 0; e < 8; ++e) {
      float om = __shfl_xor(mv[e], off, 16);
      float ol = __shfl_xor(lv[e], off, 16);
      float M = fmaxf(mv[e], om);
      lv[e] = lv[e] * __expf(mv[e] - M) + ol * __expf(om - M);
      mv[e] = M;
    }
  if ((lane & 15) == 0) {
    int rb = qb + ((lane >> 4) << 3);
#pragma unroll
    for (int e = 0; e < 8; ++e) {
      mrow[(size_t)b * Ss + rb + e] = mv[e];
      lrow[(size_t)b * Ss + rb + e] = lv[e];
    }
  }
}

// ---------------- spatial attention pass B: o = P*V, alpha, residual ----------------
__global__ __launch_bounds__(256) void k_attn_out(
    const unsigned short* __restrict__ qt, const unsigned short* __restrict__ kt,
    const unsigned short* __restrict__ vcm,
    const float* __restrict__ mrow, const float* __restrict__ lrow,
    const float* __restrict__ alpha_p,
    const unsigned short* __restrict__ s1t,
    unsigned short* __restrict__ sa_in /* padded feature-last */) {
  __shared__ alignas(16) unsigned short pT[32 * 32];
  int b = blockIdx.y, qb = blockIdx.x * 32;
  int tid = threadIdx.x, wave = tid >> 5, lane = tid & 31;
  int r = lane & 15, hi8 = (lane >> 4) << 3;
  int c0 = wave * 64;
  int mi = wave >> 1, si = wave & 1;

  v8f acc[2][4];
#pragma unroll
  for (int mf = 0; mf < 2; ++mf)
#pragma unroll
    for (int nf = 0; nf < 4; ++nf) acc[mf][nf] = vzero();

  v16u aq0 = frag_zero(), aq1 = frag_zero();
  float mv[8], iv[8];
#pragma unroll
  for (int e = 0; e < 8; ++e) { mv[e] = 0.f; iv[e] = 0.f; }
  if (wave < 4) {
    const unsigned short* qbse = qt + ((size_t)b * Ss + qb + mi * 16) * CI;
    aq0 = frag_ldA(qbse, CI, lane);
    aq1 = frag_ldA(qbse + 32, CI, lane);
#pragma unroll
    for (int e = 0; e < 8; ++e) {
      int rq = qb + mi * 16 + e + hi8;
      mv[e] = mrow[(size_t)b * Ss + rq];
      iv[e] = 1.f / lrow[(size_t)b * Ss + rq];
    }
  }
  for (int sb = 0; sb < Ss; sb += 32) {
    if (wave < 4) {
      const unsigned short* kb = kt + ((size_t)b * Ss + sb + si * 16) * CI;
      v16u b0 = frag_ldB(kb, CI, lane);
      v16u b1 = frag_ldB(kb + 32, CI, lane);
      v8f sc = vzero();
      sc = wmma32(aq0, b0, sc);
      sc = wmma32(aq1, b1, sc);
#pragma unroll
      for (int e = 0; e < 8; ++e) {
        int rr = mi * 16 + e + hi8, cc = si * 16 + r;
        pT[rr * 32 + cc] = f2bf(__expf(sc[e] - mv[e]) * iv[e]);
      }
    }
    __syncthreads();
    v16u a0 = frag_ldA(pT, 32, lane);
    v16u a1 = frag_ldA(pT + 16 * 32, 32, lane);
#pragma unroll
    for (int nf = 0; nf < 4; ++nf) {
      v16u bf = frag_ldB(vcm + ((size_t)b * Cc + c0 + nf * 16) * Ss + sb, Ss, lane);
      acc[0][nf] = wmma32(a0, bf, acc[0][nf]);
      acc[1][nf] = wmma32(a1, bf, acc[1][nf]);
    }
    __syncthreads();
  }
  float alpha = alpha_p[0];
#pragma unroll
  for (int mf = 0; mf < 2; ++mf)
#pragma unroll
    for (int nf = 0; nf < 4; ++nf)
#pragma unroll
      for (int e = 0; e < 8; ++e) {
        int row = qb + mf * 16 + e + hi8;
        int col = c0 + nf * 16 + r;
        float vv = alpha * acc[mf][nf][e] + bf2f(s1t[((size_t)b * Ss + row) * Cc + col]);
        sa_in[((size_t)b * SP + pad_off(row)) * Cc + col] = f2bf(vv);
      }
}

// ---------------- channel softmax: catt[i][j] = softmax_j G[i][j] ----------------
__global__ __launch_bounds__(128) void k_rowsoftmax(const float* __restrict__ G,
                                                    unsigned short* __restrict__ catt) {
  int b = blockIdx.y, i = blockIdx.x;
  const float* row = G + ((size_t)b * Cc + i) * Cc;
  int t = threadIdx.x;
  __shared__ float red[128];
  float mx = -3.0e38f;
  for (int j = t; j < Cc; j += 128) mx = fmaxf(mx, row[j]);
  red[t] = mx; __syncthreads();
  for (int o = 64; o; o >>= 1) { if (t < o) red[t] = fmaxf(red[t], red[t + o]); __syncthreads(); }
  mx = red[0]; __syncthreads();
  float s = 0.f;
  for (int j = t; j < Cc; j += 128) s += __expf(row[j] - mx);
  red[t] = s; __syncthreads();
  for (int o = 64; o; o >>= 1) { if (t < o) red[t] += red[t + o]; __syncthreads(); }
  float inv = 1.f / red[0];
  unsigned short* orow = catt + ((size_t)b * Cc + i) * Cc;
  for (int j = t; j < Cc; j += 128) orow[j] = f2bf(__expf(row[j] - mx) * inv);
}

// ---------------- host side ----------------
extern "C" void kernel_launch(void* const* d_in, const int* in_sizes, int n_in,
                              void* d_out, int out_size, void* d_ws, size_t ws_size,
                              hipStream_t stream) {
  (void)in_sizes; (void)n_in; (void)out_size; (void)ws_size;
  const float* x     = (const float*)d_in[0];
  const float* qw    = (const float*)d_in[6];
  const float* qb    = (const float*)d_in[7];
  const float* kw    = (const float*)d_in[8];
  const float* kb    = (const float*)d_in[9];
  const float* vw    = (const float*)d_in[10];
  const float* vb    = (const float*)d_in[11];
  const float* alpha = (const float*)d_in[12];
  const float* beta  = (const float*)d_in[23];
  float* out = (float*)d_out;

  char* p = (char*)d_ws;
  auto carve = [&](size_t bytes) -> void* {
    void* r = (void*)p;
    p += (bytes + 255) & ~(size_t)255;
    return r;
  };
  const size_t BSC  = (size_t)Bn * Ss * Cc;
  const size_t BSCP = (size_t)Bn * SP * Cc; // padded feature-last
  unsigned short* xt    = (unsigned short*)carve(BSCP * 2);
  unsigned short* w16   = (unsigned short*)carve((size_t)4 * 9 * Cc * Cc * 2); // sa1,sa2,ca1,ca2
  float*          biasf = (float*)carve((size_t)4 * Cc * 4);
  unsigned short* qw16  = (unsigned short*)carve((size_t)CI * Cc * 2);
  unsigned short* kw16  = (unsigned short*)carve((size_t)CI * Cc * 2);
  unsigned short* vw16  = (unsigned short*)carve((size_t)Cc * Cc * 2);
  unsigned short* s1t   = (unsigned short*)carve(BSC * 2);
  unsigned short* c1t   = (unsigned short*)carve(BSC * 2);
  unsigned short* c1cm  = (unsigned short*)carve(BSC * 2);
  unsigned short* qt    = (unsigned short*)carve((size_t)Bn * Ss * CI * 2);
  unsigned short* kt    = (unsigned short*)carve((size_t)Bn * Ss * CI * 2);
  unsigned short* vcm   = (unsigned short*)carve(BSC * 2);
  float*          mrow  = (float*)carve((size_t)Bn * Ss * 4);
  float*          lrow  = (float*)carve((size_t)Bn * Ss * 4);
  unsigned short* sain  = (unsigned short*)carve(BSCP * 2);
  float*          G     = (float*)carve((size_t)Bn * Cc * Cc * 4);
  unsigned short* catt  = (unsigned short*)carve((size_t)Bn * Cc * Cc * 2);
  unsigned short* cain  = (unsigned short*)carve(BSCP * 2);

  // zero the padded conv-input buffers (halo must be 0); capturable async memset
  hipMemsetAsync(xt,   0, BSCP * 2, stream);
  hipMemsetAsync(sain, 0, BSCP * 2, stream);
  hipMemsetAsync(cain, 0, BSCP * 2, stream);

  // --- prep: transpose+cast x; fold BN into conv weights; cast qkv weights ---
  {
    size_t tot = BSC;
    k_cast_x<<<dim3((unsigned)((tot + 255) / 256)), dim3(256), 0, stream>>>(x, xt);
  }
  const int widx[4] = {1, 13, 18, 24}; // sa_w1, sa_w2, ca_w1, ca_w2 (+g,b,m,v follow)
  for (int l = 0; l < 4; ++l) {
    const float* w_ = (const float*)d_in[widx[l] + 0];
    const float* g_ = (const float*)d_in[widx[l] + 1];
    const float* b_ = (const float*)d_in[widx[l] + 2];
    const float* m_ = (const float*)d_in[widx[l] + 3];
    const float* v_ = (const float*)d_in[widx[l] + 4];
    size_t tot = (size_t)9 * Cc * Cc;
    k_fold_w<<<dim3((unsigned)((tot + 255) / 256)), dim3(256), 0, stream>>>(
        w_, g_, v_, w16 + (size_t)l * 9 * Cc * Cc);
    k_fold_b<<<dim3(2), dim3(256), 0, stream>>>(b_, m_, g_, v_, biasf + (size_t)l * Cc);
  }
  k_cast<<<dim3((CI * Cc + 255) / 256), dim3(256), 0, stream>>>(qw, qw16, CI * Cc);
  k_cast<<<dim3((CI * Cc + 255) / 256), dim3(256), 0, stream>>>(kw, kw16, CI * Cc);
  k_cast<<<dim3((Cc * Cc + 255) / 256), dim3(256), 0, stream>>>(vw, vw16, Cc * Cc);

  // --- conv1 (both branches), fused BN+ReLU ---
  dim3 cgrid(Ss / 128, Cc / 128, Bn), cblk(256);
  k_conv_gemm<<<cgrid, cblk, 0, stream>>>(xt, w16 + (size_t)0 * 9 * Cc * Cc, biasf + 0 * Cc,
                                          s1t, nullptr, nullptr, 0);
  k_conv_gemm<<<cgrid, cblk, 0, stream>>>(xt, w16 + (size_t)2 * 9 * Cc * Cc, biasf + 2 * Cc,
                                          c1t, c1cm, nullptr, 0);

  const size_t sFT = (size_t)Ss * Cc, sPD = (size_t)SP * Cc;
  const size_t sQK = (size_t)Ss * CI, sCC = (size_t)Cc * Cc;
  // --- q, k, v projections (1x1 convs) ---
  k_gemm<<<dim3(CI / 64, Ss / 128, Bn), cblk, 0, stream>>>(
      s1t, sFT, Cc, qw16, 0, Cc, Cc, CI, qb, 1, nullptr, nullptr, 0,
      qt, sQK, 0, nullptr, 0, 0, nullptr, 0);
  k_gemm<<<dim3(CI / 64, Ss / 128, Bn), cblk, 0, stream>>>(
      s1t, sFT, Cc, kw16, 0, Cc, Cc, CI, kb, 1, nullptr, nullptr, 0,
      kt, sQK, 0, nullptr, 0, 0, nullptr, 0);
  k_gemm<<<dim3(Cc / 64, Ss / 128, Bn), cblk, 0, stream>>>(
      s1t, sFT, Cc, vw16, 0, Cc, Cc, Cc, vb, 1, nullptr, nullptr, 0,
      nullptr, 0, 0, vcm, sFT, Ss, nullptr, 0);

  // --- spatial attention (two-pass flash) -> sa_in = alpha*o + s1 (padded) ---
  k_attn_stats<<<dim3(Ss / 16, Bn), dim3(32), 0, stream>>>(qt, kt, mrow, lrow);
  k_attn_out<<<dim3(Ss / 32, Bn), dim3(256), 0, stream>>>(qt, kt, vcm, mrow, lrow,
                                                          alpha, s1t, sain);
  // --- spatial conv2 -> writes d_out ---
  k_conv_gemm<<<cgrid, cblk, 0, stream>>>(sain, w16 + (size_t)1 * 9 * Cc * Cc, biasf + 1 * Cc,
                                          nullptr, nullptr, out, 0);

  // --- channel attention: Gram, softmax, apply, residual ---
  k_gemm<<<dim3(Cc / 64, Cc / 128, Bn), cblk, 0, stream>>>(
      c1cm, sFT, Ss, c1cm, sFT, Ss, Ss, Cc, nullptr, 0, nullptr, nullptr, 0,
      nullptr, 0, 0, nullptr, 0, 0, G, sCC);
  k_rowsoftmax<<<dim3(Cc, Bn), dim3(128), 0, stream>>>(G, catt);
  k_gemm<<<dim3(Cc / 64, Ss / 128, Bn), cblk, 0, stream>>>(
      c1t, sFT, Cc, catt, sCC, Cc, Cc, Cc, nullptr, 2, beta, c1t, sFT,
      cain, sPD, 1, nullptr, 0, 0, nullptr, 0);
  // --- channel conv2 -> accumulates into d_out ---
  k_conv_gemm<<<cgrid, cblk, 0, stream>>>(cain, w16 + (size_t)3 * 9 * Cc * Cc, biasf + 3 * Cc,
                                          nullptr, nullptr, out, 1);
}
